// LinearAttention_45698452030311
// MI455X (gfx1250) — compile-verified, matrix operand order checked by reading, not run on hardware
//
#include <hip/hip_runtime.h>
#include <hip/hip_bf16.h>
#include <cmath>

#define DIMX   1024
#define INNERX 512
#define NSEQ   4096
#define NBATCH 4
#define HEADS  8
#define NPAIR  (NBATCH*HEADS)   // 32
#define ROWS   (NBATCH*NSEQ)    // 16384
#define LDQKV  (3*INNERX)       // 1536

typedef __attribute__((ext_vector_type(16))) __bf16 v16bf;
typedef __attribute__((ext_vector_type(8)))  float  v8f;
typedef __attribute__((ext_vector_type(2)))  float  v2f;

struct bf2 { __bf16 hi, lo; };

__device__ __forceinline__ bf2 splitf(float x) {
    bf2 r;
    r.hi = (__bf16)x;
    r.lo = (__bf16)(x - (float)r.hi);
    return r;
}

__device__ __forceinline__ float elu1(float x) {
    return x > 0.0f ? x + 1.0f : __expf(x);
}

// Build split-bf16 A fragment from a row-major f32 row (two runs of 8 floats).
__device__ __forceinline__ void build_afrag(const float* __restrict__ apf,
                                            v16bf& ahi, v16bf& alo) {
    float av[16];
    const float4 q0 = *(const float4*)(apf);
    const float4 q1 = *(const float4*)(apf + 4);
    const float4 q2 = *(const float4*)(apf + 16);
    const float4 q3 = *(const float4*)(apf + 20);
    av[0]=q0.x; av[1]=q0.y; av[2]=q0.z; av[3]=q0.w;
    av[4]=q1.x; av[5]=q1.y; av[6]=q1.z; av[7]=q1.w;
    av[8]=q2.x; av[9]=q2.y; av[10]=q2.z; av[11]=q2.w;
    av[12]=q3.x; av[13]=q3.y; av[14]=q3.z; av[15]=q3.w;
#pragma unroll
    for (int j = 0; j < 16; ++j) {
        bf2 s = splitf(av[j]);
        ahi[j] = s.hi;
        alo[j] = s.lo;
    }
}

// ---------------------------------------------------------------------------
// Weight pre-swizzle: row-major f32 W[K][N] -> fragment-layout split-bf16.
// Fragment index frag = nt*(K/32)+kt; element j of lane l holds
// W[k][n] with n = nt*16 + (l&15), k = kt*32 + 16*(j>>3) + 8*(l>>4) + (j&7).
// One thread per (frag, lane): 16 gathers, 2x32B contiguous stores.
// ---------------------------------------------------------------------------
__global__ __launch_bounds__(256)
void swizzleB_kernel(const float* __restrict__ W, int Kdim, int Ndim,
                     __bf16* __restrict__ Bhi, __bf16* __restrict__ Blo) {
    const int tid  = blockIdx.x * blockDim.x + threadIdx.x;
    const int lane = tid & 31;
    const int frag = tid >> 5;
    const int ktiles = Kdim >> 5;
    const int nfrag  = (Ndim >> 4) * ktiles;
    if (frag >= nfrag) return;
    const int nt = frag / ktiles;
    const int kt = frag - nt * ktiles;
    const int n  = nt * 16 + (lane & 15);
    const int kb = kt * 32 + 8 * (lane >> 4);
    v16bf hi, lo;
#pragma unroll
    for (int j = 0; j < 16; ++j) {
        const int k = kb + 16 * (j >> 3) + (j & 7);
        bf2 s = splitf(W[(size_t)k * Ndim + n]);
        hi[j] = s.hi;
        lo[j] = s.lo;
    }
    const size_t dst = (size_t)frag * 32 + lane;
    ((v16bf*)Bhi)[dst] = hi;
    ((v16bf*)Blo)[dst] = lo;
}

// ---------------------------------------------------------------------------
// Stage 1: qkv = x @ W_qkv  (M=16384, N=1536, K=1024), split-bf16 WMMA,
// pre-swizzled B; fused elu+1 on q,k; scatter to head-major [pair,n,d].
// Block = 8 waves; block tile = 128(M) x 64(N); wave = 16 x 64.
// ---------------------------------------------------------------------------
__global__ __launch_bounds__(256)
void qkv_kernel(const float* __restrict__ X,
                const __bf16* __restrict__ Whi, const __bf16* __restrict__ Wlo,
                float* __restrict__ Q, float* __restrict__ K, float* __restrict__ V) {
    const int wave = threadIdx.x >> 5;
    const int lane = threadIdx.x & 31;
    const int l15  = lane & 15;
    const int g    = lane >> 4;
    const int n0   = blockIdx.x * 64;
    const int m0   = blockIdx.y * 128 + wave * 16;
    const int ktiles = DIMX >> 5;          // 32

    v8f acc[4] = {};
    const float* arow = X + (size_t)(m0 + l15) * DIMX;
    const v16bf* Bh = (const v16bf*)Whi;
    const v16bf* Bl = (const v16bf*)Wlo;

    for (int kt = 0; kt < ktiles; ++kt) {
        v16bf ahi, alo;
        build_afrag(arow + kt * 32 + 8 * g, ahi, alo);
#pragma unroll
        for (int nn = 0; nn < 4; ++nn) {
            const size_t fi = ((size_t)(n0 / 16 + nn) * ktiles + kt) * 32 + lane;
            const v16bf bh = Bh[fi];
            const v16bf bl = Bl[fi];
            acc[nn] = __builtin_amdgcn_wmma_f32_16x16x32_bf16(false, ahi, false, bh,
                                                              (short)0, acc[nn], false, false);
            acc[nn] = __builtin_amdgcn_wmma_f32_16x16x32_bf16(false, ahi, false, bl,
                                                              (short)0, acc[nn], false, false);
            acc[nn] = __builtin_amdgcn_wmma_f32_16x16x32_bf16(false, alo, false, bh,
                                                              (short)0, acc[nn], false, false);
        }
    }

#pragma unroll
    for (int nn = 0; nn < 4; ++nn) {
        const int col   = n0 + nn * 16 + l15;
        const int which = col >> 9;          // 0=q 1=k 2=v
        const int inner = col & 511;
        const int h     = inner >> 6;
        const int d     = inner & 63;
#pragma unroll
        for (int i = 0; i < 8; ++i) {
            const int row = m0 + i + 8 * g;
            const int bi  = row >> 12;
            const int np  = row & (NSEQ - 1);
            const size_t idx = ((size_t)((bi * HEADS + h) * NSEQ + np)) * 64 + d;
            const float v = acc[nn][i];
            if (which == 0)      Q[idx] = elu1(v);
            else if (which == 1) K[idx] = elu1(v);
            else                 V[idx] = v;
        }
    }
}

// ---------------------------------------------------------------------------
// Stage 2: ctx[pair][d][e] = sum_n K[pair][n][d] * V[pair][n][e]
// fp32 WMMA 16x16x4; n split over 8 chunks with atomic f32 accumulation.
// ---------------------------------------------------------------------------
__global__ __launch_bounds__(256)
void context_kernel(const float* __restrict__ K, const float* __restrict__ V,
                    float* __restrict__ CTX) {
    const int pair  = blockIdx.x;
    const int chunk = blockIdx.y;
    const int wave  = threadIdx.x >> 5;
    const int lane  = threadIdx.x & 31;
    const int l15   = lane & 15;
    const int g     = lane >> 4;

    const float* Kp = K + (size_t)pair * NSEQ * 64;
    const float* Vp = V + (size_t)pair * NSEQ * 64;
    float* cp = CTX + (size_t)pair * 64 * 64;

#pragma unroll
    for (int tt = 0; tt < 2; ++tt) {
        const int t  = wave * 2 + tt;
        const int d0 = (t >> 2) * 16;
        const int e0 = (t & 3) * 16;
        v8f acc = {};
        const int nbeg = chunk * 512;
        for (int n = nbeg; n < nbeg + 512; n += 4) {
            const int nk = n + 2 * g;
            v2f a, b;
            a.x = Kp[(size_t)nk * 64 + d0 + l15];         // A = K^T : A[d][n]
            a.y = Kp[(size_t)(nk + 1) * 64 + d0 + l15];
            b.x = Vp[(size_t)nk * 64 + e0 + l15];         // B = V   : B[n][e]
            b.y = Vp[(size_t)(nk + 1) * 64 + e0 + l15];
            acc = __builtin_amdgcn_wmma_f32_16x16x4_f32(false, a, false, b,
                                                        (short)0, acc, false, false);
        }
#pragma unroll
        for (int i = 0; i < 8; ++i) {
            atomicAdd(&cp[(d0 + i + 8 * g) * 64 + e0 + l15], acc[i]);
        }
    }
}

// ---------------------------------------------------------------------------
// Stage 3: attn[b][n][h*64+e] = sum_d Q[pair][n][d] * ctx[pair][d][e]
// fp32 WMMA 16x16x4. Block = 8 waves -> 128 rows x 64 cols.
// ---------------------------------------------------------------------------
__global__ __launch_bounds__(256)
void apply_kernel(const float* __restrict__ Q, const float* __restrict__ CTX,
                  float* __restrict__ ATTN) {
    const int pair = blockIdx.x;
    const int wave = threadIdx.x >> 5;
    const int lane = threadIdx.x & 31;
    const int l15  = lane & 15;
    const int g    = lane >> 4;
    const int m0   = blockIdx.y * 128 + wave * 16;
    const int bi   = pair >> 3;
    const int h    = pair & 7;

    const float* Qp = Q + (size_t)pair * NSEQ * 64;
    const float* Cp = CTX + (size_t)pair * 64 * 64;

    v8f acc[4] = {};
    for (int k0 = 0; k0 < 64; k0 += 4) {
        const int kd = k0 + 2 * g;
        v2f a;
        a.x = Qp[(size_t)(m0 + l15) * 64 + kd];
        a.y = Qp[(size_t)(m0 + l15) * 64 + kd + 1];
#pragma unroll
        for (int nn = 0; nn < 4; ++nn) {
            const int e = nn * 16 + l15;
            v2f b;
            b.x = Cp[kd * 64 + e];
            b.y = Cp[(kd + 1) * 64 + e];
            acc[nn] = __builtin_amdgcn_wmma_f32_16x16x4_f32(false, a, false, b,
                                                            (short)0, acc[nn], false, false);
        }
    }
#pragma unroll
    for (int nn = 0; nn < 4; ++nn) {
        const int e = nn * 16 + l15;
#pragma unroll
        for (int i = 0; i < 8; ++i) {
            const int row = m0 + i + 8 * g;
            ATTN[((size_t)(bi * NSEQ + row)) * INNERX + h * 64 + e] = acc[nn][i];
        }
    }
}

// ---------------------------------------------------------------------------
// Stage 4: out = attn @ W_out + b_out  (M=16384, N=1024, K=512),
// split-bf16 WMMA with pre-swizzled B.
// ---------------------------------------------------------------------------
__global__ __launch_bounds__(256)
void out_kernel(const float* __restrict__ A,
                const __bf16* __restrict__ Whi, const __bf16* __restrict__ Wlo,
                const float* __restrict__ bias, float* __restrict__ OUT) {
    const int wave = threadIdx.x >> 5;
    const int lane = threadIdx.x & 31;
    const int l15  = lane & 15;
    const int g    = lane >> 4;
    const int n0   = blockIdx.x * 64;
    const int m0   = blockIdx.y * 128 + wave * 16;
    const int ktiles = INNERX >> 5;        // 16

    v8f acc[4] = {};
    const float* arow = A + (size_t)(m0 + l15) * INNERX;
    const v16bf* Bh = (const v16bf*)Whi;
    const v16bf* Bl = (const v16bf*)Wlo;

    for (int kt = 0; kt < ktiles; ++kt) {
        v16bf ahi, alo;
        build_afrag(arow + kt * 32 + 8 * g, ahi, alo);
#pragma unroll
        for (int nn = 0; nn < 4; ++nn) {
            const size_t fi = ((size_t)(n0 / 16 + nn) * ktiles + kt) * 32 + lane;
            const v16bf bh = Bh[fi];
            const v16bf bl = Bl[fi];
            acc[nn] = __builtin_amdgcn_wmma_f32_16x16x32_bf16(false, ahi, false, bh,
                                                              (short)0, acc[nn], false, false);
            acc[nn] = __builtin_amdgcn_wmma_f32_16x16x32_bf16(false, ahi, false, bl,
                                                              (short)0, acc[nn], false, false);
            acc[nn] = __builtin_amdgcn_wmma_f32_16x16x32_bf16(false, alo, false, bh,
                                                              (short)0, acc[nn], false, false);
        }
    }
#pragma unroll
    for (int nn = 0; nn < 4; ++nn) {
        const int col = n0 + nn * 16 + l15;
        const float bv = bias[col];
#pragma unroll
        for (int i = 0; i < 8; ++i) {
            const int row = m0 + i + 8 * g;
            OUT[(size_t)row * DIMX + col] = acc[nn][i] + bv;
        }
    }
}

// ---------------------------------------------------------------------------
extern "C" void kernel_launch(void* const* d_in, const int* in_sizes, int n_in,
                              void* d_out, int out_size, void* d_ws, size_t ws_size,
                              hipStream_t stream) {
    const float* x    = (const float*)d_in[0];
    const float* Wqkv = (const float*)d_in[1];
    const float* Wout = (const float*)d_in[2];
    const float* bout = (const float*)d_in[3];
    float* out = (float*)d_out;

    const size_t per = (size_t)NPAIR * NSEQ * 64;     // 8,388,608 floats
    char* w = (char*)d_ws;
    float* Q    = (float*)w;  w += per * sizeof(float);
    float* K    = (float*)w;  w += per * sizeof(float);
    float* V    = (float*)w;  w += per * sizeof(float);
    float* CTX  = (float*)w;  w += (size_t)NPAIR * 64 * 64 * sizeof(float);
    float* ATTN = (float*)w;  w += (size_t)ROWS * INNERX * sizeof(float);
    __bf16* WqkvHi = (__bf16*)w; w += (size_t)DIMX * LDQKV * sizeof(__bf16);
    __bf16* WqkvLo = (__bf16*)w; w += (size_t)DIMX * LDQKV * sizeof(__bf16);
    __bf16* WoutHi = (__bf16*)w; w += (size_t)INNERX * DIMX * sizeof(__bf16);
    __bf16* WoutLo = (__bf16*)w; w += (size_t)INNERX * DIMX * sizeof(__bf16);

    (void)hipMemsetAsync(CTX, 0, (size_t)NPAIR * 64 * 64 * sizeof(float), stream);

    {   // W_qkv: (1536/16)*(1024/32) = 3072 frags * 32 lanes = 98304 threads
        const int threads = (LDQKV / 16) * (DIMX / 32) * 32;
        swizzleB_kernel<<<(threads + 255) / 256, 256, 0, stream>>>(
            Wqkv, DIMX, LDQKV, WqkvHi, WqkvLo);
    }
    {   // W_out: (1024/16)*(512/32) = 1024 frags * 32 lanes = 32768 threads
        const int threads = (DIMX / 16) * (INNERX / 32) * 32;
        swizzleB_kernel<<<(threads + 255) / 256, 256, 0, stream>>>(
            Wout, INNERX, DIMX, WoutHi, WoutLo);
    }

    qkv_kernel<<<dim3(LDQKV / 64, ROWS / 128), 256, 0, stream>>>(x, WqkvHi, WqkvLo, Q, K, V);
    context_kernel<<<dim3(NPAIR, NSEQ / 512), 256, 0, stream>>>(K, V, CTX);
    apply_kernel<<<dim3(NPAIR, NSEQ / 128), 256, 0, stream>>>(Q, CTX, ATTN);
    out_kernel<<<dim3(DIMX / 64, ROWS / 128), 256, 0, stream>>>(ATTN, Wout ? WoutHi : WoutHi, WoutLo, bout, out);
}
